// VQVAE_59416577573490
// MI455X (gfx1250) — matrix operand order repaired and checked by reading, not compiled, not run on hardware
//
#include <hip/hip_runtime.h>
#include <hip/hip_bf16.h>

typedef _Float16 h16;
typedef __attribute__((ext_vector_type(8)))  _Float16 v8h;
typedef __attribute__((ext_vector_type(16))) _Float16 v16h;
typedef __attribute__((ext_vector_type(8)))  float    v8f;

// ---------------------------------------------------------------------------
// Generic implicit-GEMM convolution using V_WMMA_F32_16X16X32_F16.
//   - activations: NHWC f16, channel stride Cpad (power of two)
//   - weights: [CoutPad][Kpad] f16, K = (r*kw+q)*Cpad + c
//   - phase >= 0: ConvTranspose2d k4/s2/p1 phase decomposition (K = (a*2+b)*Cpad+c)
//   - block: 256 threads = 8 waves; tile = 64 pixels x 32 couts; A staged in LDS
//     64 K-wide per barrier round (two WMMA per round), 32-wide remainder.
//   - all divisors are powers of two -> shift/mask only (no integer division
//     except tap/kw once per chunk).
// ---------------------------------------------------------------------------
struct ConvP {
  const h16*   in;
  const h16*   wt;
  const float* bias;
  const h16*   resid;   // residual add (NHWC, stride CoutPad), nullable
  void*        out;     // h16* (NHWC) or float* (NCHW f32) per nchw_out
  int N, Hin, Win, Cpad;
  int cshift;           // log2(Cpad)
  int Hp, Wp;           // pixel-space dims (pow2)
  int wpshift, hwpshift;// log2(Wp), log2(Hp*Wp)
  int Hout, Wout;       // true output spatial dims
  int Cout, CoutPad;    // real / padded (mult of 32) output channels
  int kh, kw, stride, pad;
  int Kpad;             // GEMM K (multiple of 32; multiple of 64 except K=32)
  int phase;            // -1 normal; else phy*2+phx
  int relu_in, relu_out, nchw_out;
};

#define LDS_STRIDE 72   // 64 K halfs + 8 pad (144B rows, 16B aligned)

__global__ __launch_bounds__(256) void k_conv_wmma(ConvP p) {
  __shared__ __align__(16) h16 At[64 * LDS_STRIDE];

  const int tid  = threadIdx.x;
  const int lane = tid & 31;
  const int wave = tid >> 5;
  const int wm   = wave >> 1;        // pixel-tile row of this wave (0..3)
  const int wn   = wave & 1;         // cout-tile col of this wave (0..1)
  const int lm   = lane & 15;
  const int lh   = lane >> 4;        // lane half

  const int pixBase = blockIdx.x * 64;
  const int cnBase  = blockIdx.y * 32;
  const int wpmask  = p.Wp - 1;
  const int hwmask  = (1 << p.hwpshift) - 1;
  const int cmask   = p.Cpad - 1;

  int phy = 0, phx = 0;
  if (p.phase >= 0) { phy = p.phase >> 1; phx = p.phase & 1; }

  // ---- per-thread A-tile loader coordinates (fixed pixel, fixed k slice) ----
  const int pl   = tid >> 2;          // local pixel 0..63
  const int kloc = (tid & 3) * 8;     // 0,8,16,24
  const int pg   = pixBase + pl;
  const int ln   = pg >> p.hwpshift;
  const int lrem = pg & hwmask;
  const int py   = lrem >> p.wpshift;
  const int px   = lrem & wpmask;
  const bool vecC = (p.Cpad & 7) == 0;

  // stage 8 halfs of im2col row `pl`, K slice [kbase, kbase+8) -> At[pl][ldk..]
  auto stage8 = [&](int kbase, int ldk) {
    if (vecC) {
      const int tap = kbase >> p.cshift;
      const int c   = kbase & cmask;
      int iy, ix;
      if (p.phase < 0) {
        const int r = tap / p.kw, q = tap - r * p.kw;
        iy = py * p.stride - p.pad + r;
        ix = px * p.stride - p.pad + q;
      } else {
        iy = py + phy - (tap >> 1);
        ix = px + phx - (tap & 1);
      }
      v8h d = {};
      if (iy >= 0 && iy < p.Hin && ix >= 0 && ix < p.Win)
        d = *(const v8h*)(p.in + (((ln * p.Hin + iy) * p.Win + ix) << p.cshift) + c);
      if (p.relu_in) {
#pragma unroll
        for (int j = 0; j < 8; ++j) d[j] = d[j] > (h16)0 ? d[j] : (h16)0;
      }
      *(v8h*)&At[pl * LDS_STRIDE + ldk] = d;
    } else {
      // scalar fallback (first layer, Cpad==4)
#pragma unroll
      for (int j = 0; j < 8; ++j) {
        const int k   = kbase + j;
        const int tap = k >> p.cshift, c = k & cmask;
        const int r   = tap / p.kw, q = tap - r * p.kw;
        const int iy  = py * p.stride - p.pad + r;
        const int ix  = px * p.stride - p.pad + q;
        h16 v = (h16)0;
        if (iy >= 0 && iy < p.Hin && ix >= 0 && ix < p.Win)
          v = p.in[(((ln * p.Hin + iy) * p.Win + ix) << p.cshift) + c];
        if (p.relu_in) v = v > (h16)0 ? v : (h16)0;
        At[pl * LDS_STRIDE + ldk + j] = v;
      }
    }
  };

  // A fragment (ISA 16-bit A 16x32 layout) from LDS at K offset `koff`
  auto afrag = [&](int koff) -> v16h {
    const h16* arow = &At[(wm * 16 + lm) * LDS_STRIDE + koff + lh * 8];
    const v8h a0 = *(const v8h*)(arow);
    const v8h a1 = *(const v8h*)(arow + 16);
    v16h av = {};
#pragma unroll
    for (int j = 0; j < 8; ++j) { av[j] = a0[j]; av[j + 8] = a1[j]; }
    return av;
  };
  // B fragment: lane column = cout, K = 8*lh + j and 16 + 8*lh + j (matches A layout)
  auto bfrag = [&](int kglob) -> v16h {
    const h16* wrow = p.wt + (cnBase + wn * 16 + lm) * p.Kpad + kglob + lh * 8;
    const v8h b0 = *(const v8h*)(wrow);
    const v8h b1 = *(const v8h*)(wrow + 16);
    v16h bv = {};
#pragma unroll
    for (int j = 0; j < 8; ++j) { bv[j] = b0[j]; bv[j + 8] = b1[j]; }
    return bv;
  };

  v8f acc = {};
  const int nk2 = p.Kpad >> 6;          // 64-wide rounds
  for (int kk = 0; kk < nk2; ++kk) {
    __syncthreads();
    stage8(kk * 64 + kloc,      kloc);
    stage8(kk * 64 + 32 + kloc, 32 + kloc);
    __syncthreads();
#pragma unroll
    for (int sub = 0; sub < 2; ++sub) {
      acc = __builtin_amdgcn_wmma_f32_16x16x32_f16(false, afrag(sub * 32), false,
                                                   bfrag(kk * 64 + sub * 32),
                                                   (short)0, acc, false, false);
    }
  }
  if (p.Kpad & 32) {                    // 32-wide remainder (K=32 layers)
    __syncthreads();
    stage8(nk2 * 64 + kloc, kloc);
    __syncthreads();
    acc = __builtin_amdgcn_wmma_f32_16x16x32_f16(false, afrag(0), false,
                                                 bfrag(nk2 * 64),
                                                 (short)0, acc, false, false);
  }

  // ---- epilogue: bias, residual, relu, store ----
  const int  cout = cnBase + wn * 16 + lm;
  const float bv  = (p.bias && cout < p.Cout) ? p.bias[cout] : 0.f;
#pragma unroll
  for (int r = 0; r < 8; ++r) {
    const int plo = wm * 16 + lh * 8 + r;   // C/D layout: VGPR r, lane half
    const int pp  = pixBase + plo;
    float v = acc[r] + bv;
    if (p.resid) v += (float)p.resid[pp * p.CoutPad + cout];
    if (p.relu_out) v = fmaxf(v, 0.f);
    if (!p.nchw_out) {
      if (p.phase < 0) {
        ((h16*)p.out)[pp * p.CoutPad + cout] = (h16)v;
      } else {
        const int nn = pp >> p.hwpshift, rr = pp & hwmask;
        const int oy = 2 * (rr >> p.wpshift) + phy;
        const int ox = 2 * (rr & wpmask) + phx;
        ((h16*)p.out)[((nn * p.Hout + oy) * p.Wout + ox) * p.CoutPad + cout] = (h16)v;
      }
    } else if (cout < p.Cout) {
      const int nn = pp >> p.hwpshift, rr = pp & hwmask;
      int oy = rr >> p.wpshift, ox = rr & wpmask;
      if (p.phase >= 0) { oy = 2 * oy + phy; ox = 2 * ox + phx; }
      ((float*)p.out)[((nn * p.Cout + cout) * p.Hout + oy) * p.Wout + ox] = v;
    }
  }
}

// ---------------------------------------------------------------------------
// Weight preparation: OIHW fp32 -> [CoutPad][Kpad] f16, K = (r*kw+q)*Cpad+c
// ---------------------------------------------------------------------------
__global__ void k_prep_w(const float* src, h16* dst, int Cout, int Cin,
                         int kh, int kw, int Cpad, int CoutPad) {
  const int Kpad = kh * kw * Cpad;
  const int idx = blockIdx.x * 256 + threadIdx.x;
  if (idx >= CoutPad * Kpad) return;
  const int cout = idx / Kpad, k = idx % Kpad;
  const int tap = k / Cpad, c = k - tap * Cpad;
  const int r = tap / kw, q = tap - r * kw;
  float v = 0.f;
  if (cout < Cout && c < Cin)
    v = src[((cout * Cin + c) * kh + r) * kw + q];
  dst[idx] = (h16)v;
}

// ConvTranspose2d k4/s2/p1 phase weights: src (Cin,Cout,4,4) fp32 ->
// [CoutPad][4*Cin] f16, K=(a*2+b)*Cin+c, tap value = w[c][cout][1-phy+2a][1-phx+2b]
__global__ void k_prep_wt(const float* src, h16* dst, int Cin, int Cout,
                          int CoutPad, int phy, int phx) {
  const int Kpad = 4 * Cin;
  const int idx = blockIdx.x * 256 + threadIdx.x;
  if (idx >= CoutPad * Kpad) return;
  const int cout = idx / Kpad, k = idx % Kpad;
  const int tap = k / Cin, c = k - tap * Cin;
  const int a = tap >> 1, b = tap & 1;
  float v = 0.f;
  if (cout < Cout)
    v = src[((c * Cout + cout) * 4 + (1 - phy + 2 * a)) * 4 + (1 - phx + 2 * b)];
  dst[idx] = (h16)v;
}

// NCHW fp32 input image -> NHWC f16 padded to 4 channels
__global__ void k_cast_x(const float* x, h16* xp) {
  const int i = blockIdx.x * 256 + threadIdx.x;   // over 16*256*256*4
  const int c = i & 3;
  const int pix = i >> 2;
  const int n = pix >> 16, rem = pix & 65535;
  const int y = rem >> 8, xx = rem & 255;
  float v = 0.f;
  if (c < 3) v = x[((n * 3 + c) * 256 + y) * 256 + xx];
  xp[i] = (h16)v;
}

// codebook (64,512) fp32 -> cbT[512][64] f16 + cnorm[512] f32
__global__ void k_prep_cb(const float* cb, h16* cbT, float* cnorm) {
  const int k = blockIdx.x * 256 + threadIdx.x;
  if (k >= 512) return;
  float s = 0.f;
  for (int d = 0; d < 64; ++d) {
    const float v = cb[d * 512 + k];
    s += v * v;
    cbT[k * 64 + d] = (h16)v;
  }
  cnorm[k] = s;
}

// Vector quantizer: per pixel argmin_k(||c_k||^2 - 2 z.c_k), gather code row.
__global__ __launch_bounds__(256) void k_vq(const h16* __restrict__ z,
                                            const h16* __restrict__ cbT,
                                            const float* __restrict__ cnorm,
                                            h16* __restrict__ zq, int npix) {
  const int pix = blockIdx.x * 256 + threadIdx.x;
  if (pix >= npix) return;
  float zf[64];
#pragma unroll
  for (int i = 0; i < 8; ++i) {
    const v8h d = *(const v8h*)(z + pix * 64 + i * 8);
#pragma unroll
    for (int j = 0; j < 8; ++j) zf[i * 8 + j] = (float)d[j];
  }
  float best = 3.4e38f; int bi = 0;
  for (int k = 0; k < 512; ++k) {
    float dot = 0.f;
#pragma unroll
    for (int i = 0; i < 8; ++i) {
      const v8h d = *(const v8h*)(cbT + k * 64 + i * 8);
#pragma unroll
      for (int j = 0; j < 8; ++j) dot += zf[i * 8 + j] * (float)d[j];
    }
    const float dist = cnorm[k] - 2.f * dot;
    if (dist < best) { best = dist; bi = k; }
  }
#pragma unroll
  for (int i = 0; i < 8; ++i)
    *(v8h*)(zq + pix * 64 + i * 8) = *(const v8h*)(cbT + bi * 64 + i * 8);
}

// ---------------------------------------------------------------------------
extern "C" void kernel_launch(void* const* d_in, const int* in_sizes, int n_in,
                              void* d_out, int out_size, void* d_ws, size_t ws_size,
                              hipStream_t stream) {
  (void)in_sizes; (void)n_in; (void)out_size; (void)ws_size;
  const float* x          = (const float*)d_in[0];
  const float* enc_w0     = (const float*)d_in[1];
  const float* enc_b0     = (const float*)d_in[2];
  const float* enc_w1     = (const float*)d_in[3];
  const float* enc_b1     = (const float*)d_in[4];
  const float* enc_wf     = (const float*)d_in[5];
  const float* enc_bf     = (const float*)d_in[6];
  const float* enc_res_w1 = (const float*)d_in[7];
  const float* enc_res_b1 = (const float*)d_in[8];
  const float* enc_res_w2 = (const float*)d_in[9];
  const float* enc_res_b2 = (const float*)d_in[10];
  const float* pre_w      = (const float*)d_in[11];
  const float* pre_b      = (const float*)d_in[12];
  const float* codebook   = (const float*)d_in[13];
  const float* dec_w      = (const float*)d_in[14];
  const float* dec_b      = (const float*)d_in[15];
  const float* dec_res_w1 = (const float*)d_in[16];
  const float* dec_res_b1 = (const float*)d_in[17];
  const float* dec_res_w2 = (const float*)d_in[18];
  const float* dec_res_b2 = (const float*)d_in[19];
  const float* up_w0      = (const float*)d_in[20];
  const float* up_b0      = (const float*)d_in[21];
  const float* up_w1      = (const float*)d_in[22];
  const float* up_b1      = (const float*)d_in[23];

  size_t off = 0;
  auto alloc = [&](size_t bytes) -> void* {
    void* pp = (char*)d_ws + off;
    off += (bytes + 255) & ~(size_t)255;
    return pp;
  };

  // --- persistent prepped weights (f16) ---
  h16* w_enc0 = (h16*)alloc((size_t)64 * 64 * 2);       // k4, Cpad4  -> K 64
  h16* w_enc1 = (h16*)alloc((size_t)128 * 1024 * 2);    // k4, Cpad64 -> K 1024
  h16* w_encf = (h16*)alloc((size_t)128 * 1152 * 2);    // k3, Cpad128-> K 1152
  h16* w_eres1[2], *w_eres2[2], *w_dres1[2], *w_dres2[2];
  for (int i = 0; i < 2; ++i) {
    w_eres1[i] = (h16*)alloc((size_t)32 * 1152 * 2);
    w_eres2[i] = (h16*)alloc((size_t)128 * 32 * 2);
    w_dres1[i] = (h16*)alloc((size_t)32 * 1152 * 2);
    w_dres2[i] = (h16*)alloc((size_t)128 * 32 * 2);
  }
  h16* w_pre = (h16*)alloc((size_t)64 * 128 * 2);
  h16* w_dec = (h16*)alloc((size_t)128 * 576 * 2);
  h16* w_up0[4], *w_up1[4];
  for (int ph = 0; ph < 4; ++ph) {
    w_up0[ph] = (h16*)alloc((size_t)64 * 512 * 2);      // phase K = 4*128
    w_up1[ph] = (h16*)alloc((size_t)32 * 256 * 2);      // phase K = 4*64
  }
  h16*   cbT   = (h16*)alloc((size_t)512 * 64 * 2);
  float* cnorm = (float*)alloc((size_t)512 * 4);

  // --- activation regions (f16 NHWC), reused across the graph ---
  char* R0 = (char*)alloc(33554432);   // h0 -> zq -> up0 out
  char* R1 = (char*)alloc(33554432);   // h1 -> hd
  char* R2 = (char*)alloc(16777216);   // x_pad -> h2
  char* R3 = (char*)alloc(8388608);    // z
  char* R4 = (char*)alloc(4194304);    // res-block temp t
  h16* x_pad = (h16*)R2;  h16* h0 = (h16*)R0;  h16* h1 = (h16*)R1;
  h16* h2 = (h16*)R2;     h16* tb = (h16*)R4;  h16* z  = (h16*)R3;
  h16* zq = (h16*)R0;     h16* hd = (h16*)R1;  h16* up0o = (h16*)R0;

  auto prep = [&](const float* src, h16* dst, int Cout, int Cin, int kh, int kw,
                  int Cpad, int CoutPad) {
    const int tot = CoutPad * kh * kw * Cpad;
    k_prep_w<<<(tot + 255) / 256, 256, 0, stream>>>(src, dst, Cout, Cin, kh, kw, Cpad, CoutPad);
  };
  auto prepT = [&](const float* src, h16* dst, int Cin, int Cout, int CoutPad, int ph) {
    const int tot = CoutPad * 4 * Cin;
    k_prep_wt<<<(tot + 255) / 256, 256, 0, stream>>>(src, dst, Cin, Cout, CoutPad, ph >> 1, ph & 1);
  };
  auto ctz = [](int v) { int s = 0; while (!(v & 1)) { v >>= 1; ++s; } return s; };
  auto conv = [&](const h16* in, const h16* wt, const float* bias, const h16* resid,
                  void* out, int N, int Hin, int Win, int Cpad,
                  int Hout, int Wout, int Cout, int CoutPad,
                  int kh, int kw, int stride, int pad, int phase,
                  int relu_in, int relu_out, int nchw) {
    ConvP p;
    p.in = in; p.wt = wt; p.bias = bias; p.resid = resid; p.out = out;
    p.N = N; p.Hin = Hin; p.Win = Win; p.Cpad = Cpad; p.cshift = ctz(Cpad);
    p.Hout = Hout; p.Wout = Wout; p.Cout = Cout; p.CoutPad = CoutPad;
    p.kh = kh; p.kw = kw; p.stride = stride; p.pad = pad; p.phase = phase;
    p.Hp = (phase >= 0) ? Hout / 2 : Hout;
    p.Wp = (phase >= 0) ? Wout / 2 : Wout;
    p.wpshift = ctz(p.Wp); p.hwpshift = ctz(p.Hp * p.Wp);
    p.Kpad = (phase >= 0) ? 4 * Cpad : kh * kw * Cpad;
    p.relu_in = relu_in; p.relu_out = relu_out; p.nchw_out = nchw;
    const int P = N * p.Hp * p.Wp;
    dim3 grid((unsigned)(P / 64), (unsigned)(CoutPad / 32));
    k_conv_wmma<<<grid, 256, 0, stream>>>(p);
  };

  // --- weight/codebook/input prep ---
  prep(enc_w0, w_enc0, 64, 3, 4, 4, 4, 64);
  prep(enc_w1, w_enc1, 128, 64, 4, 4, 64, 128);
  prep(enc_wf, w_encf, 128, 128, 3, 3, 128, 128);
  for (int i = 0; i < 2; ++i) {
    prep(enc_res_w1 + i * 32 * 128 * 9, w_eres1[i], 32, 128, 3, 3, 128, 32);
    prep(enc_res_w2 + i * 128 * 32,     w_eres2[i], 128, 32, 1, 1, 32, 128);
    prep(dec_res_w1 + i * 32 * 128 * 9, w_dres1[i], 32, 128, 3, 3, 128, 32);
    prep(dec_res_w2 + i * 128 * 32,     w_dres2[i], 128, 32, 1, 1, 32, 128);
  }
  prep(pre_w, w_pre, 64, 128, 1, 1, 128, 64);
  prep(dec_w, w_dec, 128, 64, 3, 3, 64, 128);
  for (int ph = 0; ph < 4; ++ph) {
    prepT(up_w0, w_up0[ph], 128, 64, 64, ph);
    prepT(up_w1, w_up1[ph], 64, 3, 32, ph);
  }
  k_prep_cb<<<2, 256, 0, stream>>>(codebook, cbT, cnorm);
  k_cast_x<<<16384, 256, 0, stream>>>(x, x_pad);

  // --- encoder ---
  conv(x_pad, w_enc0, enc_b0, nullptr, h0, 16, 256, 256, 4, 128, 128, 64, 64, 4, 4, 2, 1, -1, 0, 1, 0);
  conv(h0, w_enc1, enc_b1, nullptr, h1, 16, 128, 128, 64, 64, 64, 128, 128, 4, 4, 2, 1, -1, 0, 1, 0);
  conv(h1, w_encf, enc_bf, nullptr, h2, 16, 64, 64, 128, 64, 64, 128, 128, 3, 3, 1, 1, -1, 0, 0, 0);
  for (int i = 0; i < 2; ++i) {
    conv(h2, w_eres1[i], enc_res_b1 + i * 32, nullptr, tb, 16, 64, 64, 128, 64, 64, 32, 32, 3, 3, 1, 1, -1, 1, 0, 0);
    conv(tb, w_eres2[i], enc_res_b2 + i * 128, h2, h2, 16, 64, 64, 32, 64, 64, 128, 128, 1, 1, 1, 0, -1, 1, 0, 0);
  }
  conv(h2, w_pre, pre_b, nullptr, z, 16, 64, 64, 128, 64, 64, 64, 64, 1, 1, 1, 0, -1, 1, 0, 0);

  // --- vector quantizer (straight-through forward: zq = gathered code) ---
  k_vq<<<65536 / 256, 256, 0, stream>>>(z, cbT, cnorm, zq, 65536);

  // --- decoder ---
  conv(zq, w_dec, dec_b, nullptr, hd, 16, 64, 64, 64, 64, 64, 128, 128, 3, 3, 1, 1, -1, 0, 0, 0);
  for (int i = 0; i < 2; ++i) {
    conv(hd, w_dres1[i], dec_res_b1 + i * 32, nullptr, tb, 16, 64, 64, 128, 64, 64, 32, 32, 3, 3, 1, 1, -1, 1, 0, 0);
    conv(tb, w_dres2[i], dec_res_b2 + i * 128, hd, hd, 16, 64, 64, 32, 64, 64, 128, 128, 1, 1, 1, 0, -1, 1, 0, 0);
  }
  for (int ph = 0; ph < 4; ++ph)   // ConvTranspose 128->64 (relu_in = res-stack relu)
    conv(hd, w_up0[ph], up_b0, nullptr, up0o, 16, 64, 64, 128, 128, 128, 64, 64, 4, 4, 2, 1, ph, 1, 1, 0);
  for (int ph = 0; ph < 4; ++ph)   // ConvTranspose 64->3, final NCHW f32 output
    conv(up0o, w_up1[ph], up_b1, nullptr, d_out, 16, 128, 128, 64, 256, 256, 3, 32, 4, 4, 2, 1, ph, 0, 0, 1);
}
